// TFT_50843822850618
// MI455X (gfx1250) — matrix operand order repaired and checked by reading, not compiled, not run on hardware
//
#include <hip/hip_runtime.h>

// ---------------- constants (TFT shapes) ----------------
static constexpr int  kB     = 128;
static constexpr int  kTH    = 336;
static constexpr int  kTF    = 48;
static constexpr int  kT     = 384;     // kTH + kTF
static constexpr int  kNH    = 8;
static constexpr int  kNF    = 4;
static constexpr int  kH     = 256;
static constexpr int  kHEADS = 4;
static constexpr int  kDK    = 64;
static constexpr long kBTH   = (long)kB * kTH;   // 43008
static constexpr long kBTF   = (long)kB * kTF;   // 6144
static constexpr long kBT    = (long)kB * kT;    // 49152

typedef __attribute__((ext_vector_type(16))) __bf16 v16bf;
typedef __attribute__((ext_vector_type(8)))  __bf16 v8bf;
typedef __attribute__((ext_vector_type(8)))  float  v8f;

union Pk8 { __bf16 h[8]; uint4 v; };

__device__ __forceinline__ float sigf(float x) { return 1.0f / (1.0f + __expf(-x)); }

// ---------------------------------------------------------------------------
// Generic batched GEMM:  C[z] = act( scale * A[z] @ B[z] + bias[z] (+/-* add) )
//   act: 0=none, 1=ELU, 2=sigmoid, 3=sigmoid(.)*add  (GLU fusion)
//   addRowDiv: add-source row index = gm / addRowDiv  (ctx broadcast over time)
//   aMod/bMod/addMod: batch offset = (aMod ? z % aMod : z) * stride
//   transB: B accessed as [N,K] row-major (for Q @ K^T)
// Tile: 128(M)x64(N) per 256-thread block, K-step 32; each wave computes a
// 32x32 sub-tile = 4 x V_WMMA_F32_16X16X32_BF16 per K-step (A/B fragments each
// reused twice). f32 -> bf16 staged in LDS with b128 global loads + packed 16B
// LDS stores; B kept N-major in LDS so fragments are contiguous ds_load_b128
// pairs. Fragment layouts per CDNA5 ISA 7.12.2 (wave32). Edge tiles use a
// branchless clamped-address + mask-multiply path.
// ---------------------------------------------------------------------------
__global__ __launch_bounds__(256) void gemm_wmma(
    const float* __restrict__ A, const float* __restrict__ Bm,
    const float* __restrict__ bias, const float* __restrict__ Add,
    float* __restrict__ C,
    int M, int N, int K, int lda, int ldb, int ldc, int ldadd,
    long sA, long sB, long sC, long sAdd,
    int aMod, int bMod, int addMod, int biasStride,
    int transB, int act, float scale, int addRowDiv)
{
  __shared__ __bf16 As [128][40];  // [m][k]  row-major, pitch 80B (16B-multiple)
  __shared__ __bf16 BsT[64][40];   // [n][k]  N-major (pre-transposed for frags)

  const int tid  = threadIdx.x;
  const int lane = tid & 31;
  const int w    = tid >> 5;           // 8 waves: (w&3)=32-row subtile, (w>>2)=N half
  const int z    = blockIdx.z;
  const int m0   = blockIdx.y << 7;    // 128-row tiles
  const int n0   = blockIdx.x << 6;    // 64-col tiles

  const float* Ab = A  + (long)(aMod ? (z % aMod) : z) * sA;
  const float* Bb = Bm + (long)(bMod ? (z % bMod) : z) * sB;
  float*       Cb = C  + (long)z * sC;
  const float* Db = Add  ? (Add  + (long)(addMod ? (z % addMod) : z) * sAdd) : nullptr;
  const float* bb = bias ? (bias + (long)z * biasStride) : nullptr;

  v8f acc00 = {}, acc01 = {}, acc10 = {}, acc11 = {};

  const int arow = tid >> 2;           // 0..63 (M, x2 row blocks)
  const int acol = (tid & 3) << 3;     // 0,8,16,24 (K)
  const int bkN  = tid >> 3;           // 0..31 (K)    non-transB mapping
  const int bnN  = (tid & 7) << 3;     // 0..56 (N)
  const int bnT  = tid >> 2;           // 0..63 (N)    transB mapping
  const int bkT  = (tid & 3) << 3;     // 0,8,16,24 (K)

  const bool kFull = (K & 31) == 0;
  const bool fullA = kFull && (m0 + 128 <= M);
  const bool fullB = kFull && (n0 + 64 <= N);

  for (int k0 = 0; k0 < K; k0 += 32) {
    // ---- stage A tile: As[m][k], f32 -> bf16, 2 row-blocks per thread ----
    if (fullA) {
      #pragma unroll
      for (int rr = 0; rr < 2; ++rr) {
        const float* src = Ab + (long)(m0 + arow + 64 * rr) * lda + k0 + acol;
        const float4 x0 = *(const float4*)(src);
        const float4 x1 = *(const float4*)(src + 4);
        Pk8 p;
        p.h[0] = (__bf16)x0.x; p.h[1] = (__bf16)x0.y;
        p.h[2] = (__bf16)x0.z; p.h[3] = (__bf16)x0.w;
        p.h[4] = (__bf16)x1.x; p.h[5] = (__bf16)x1.y;
        p.h[6] = (__bf16)x1.z; p.h[7] = (__bf16)x1.w;
        *(uint4*)&As[arow + 64 * rr][acol] = p.v;
      }
    } else {
      #pragma unroll
      for (int rr = 0; rr < 2; ++rr) {
        const int gm = m0 + arow + 64 * rr;
        const int cm = (gm < M) ? gm : 0;
        const float rowMask = (gm < M) ? 1.0f : 0.0f;
        const float* rowp = Ab + (long)cm * lda;
        #pragma unroll
        for (int j = 0; j < 8; ++j) {
          const int gk = k0 + acol + j;
          const int ck = (gk < K) ? gk : 0;
          const float msk = (gk < K) ? rowMask : 0.0f;
          As[arow + 64 * rr][acol + j] = (__bf16)(rowp[ck] * msk);
        }
      }
    }
    // ---- stage B tile into BsT[n][k] ----
    if (!transB) {
      if (fullB) {
        const float* src = Bb + (long)(k0 + bkN) * ldb + n0 + bnN;
        const float4 x0 = *(const float4*)(src);
        const float4 x1 = *(const float4*)(src + 4);
        float vv[8] = {x0.x, x0.y, x0.z, x0.w, x1.x, x1.y, x1.z, x1.w};
        #pragma unroll
        for (int j = 0; j < 8; ++j) BsT[bnN + j][bkN] = (__bf16)vv[j];
      } else {
        const int gk = k0 + bkN;
        const int ck = (gk < K) ? gk : 0;
        const float kMask = (gk < K) ? 1.0f : 0.0f;
        const float* rowp = Bb + (long)ck * ldb;
        #pragma unroll
        for (int j = 0; j < 8; ++j) {
          const int gn = n0 + bnN + j;
          const int cn = (gn < N) ? gn : 0;
          const float msk = (gn < N) ? kMask : 0.0f;
          BsT[bnN + j][bkN] = (__bf16)(rowp[cn] * msk);
        }
      }
    } else {
      if (fullB) {
        const float* src = Bb + (long)(n0 + bnT) * ldb + k0 + bkT;
        const float4 x0 = *(const float4*)(src);
        const float4 x1 = *(const float4*)(src + 4);
        Pk8 p;
        p.h[0] = (__bf16)x0.x; p.h[1] = (__bf16)x0.y;
        p.h[2] = (__bf16)x0.z; p.h[3] = (__bf16)x0.w;
        p.h[4] = (__bf16)x1.x; p.h[5] = (__bf16)x1.y;
        p.h[6] = (__bf16)x1.z; p.h[7] = (__bf16)x1.w;
        *(uint4*)&BsT[bnT][bkT] = p.v;
      } else {
        const int gn = n0 + bnT;
        const int cn = (gn < N) ? gn : 0;
        const float nMask = (gn < N) ? 1.0f : 0.0f;
        const float* rowp = Bb + (long)cn * ldb;
        #pragma unroll
        for (int j = 0; j < 8; ++j) {
          const int gk = k0 + bkT + j;
          const int ck = (gk < K) ? gk : 0;
          const float msk = (gk < K) ? nMask : 0.0f;
          BsT[bnT][bkT + j] = (__bf16)(rowp[ck] * msk);
        }
      }
    }
    if (k0 + 32 < K)  // emits global_prefetch_b8
      __builtin_prefetch(Ab + (long)(m0 + arow) * lda + (k0 + 32 + acol), 0, 0);
    __syncthreads();

    // ---- A fragments (16x32 each): lane = M row; K halves per ISA layout ----
    const int mrow = ((w & 3) << 5) | (lane & 15);     // subtile base + row
    const int kbA  = (lane < 16) ? 0 : 8;
    const v8bf a00 = *(const v8bf*)&As[mrow][kbA];
    const v8bf a01 = *(const v8bf*)&As[mrow][kbA + 16];
    const v16bf af0 = __builtin_shufflevector(a00, a01,
        0, 1, 2, 3, 4, 5, 6, 7, 8, 9, 10, 11, 12, 13, 14, 15);
    const v8bf a10 = *(const v8bf*)&As[mrow + 16][kbA];
    const v8bf a11 = *(const v8bf*)&As[mrow + 16][kbA + 16];
    const v16bf af1 = __builtin_shufflevector(a10, a11,
        0, 1, 2, 3, 4, 5, 6, 7, 8, 9, 10, 11, 12, 13, 14, 15);

    // ---- B fragments (32x16 each): lane = N column; K blocked by lane half ----
    const int nbase = (w >> 2) << 5;
    const int kbB   = (lane < 16) ? 0 : 8;
    const int nc    = lane & 15;
    const v8bf b0l = *(const v8bf*)&BsT[nbase + nc][kbB];
    const v8bf b0h = *(const v8bf*)&BsT[nbase + nc][kbB + 16];
    const v16bf bf0 = __builtin_shufflevector(b0l, b0h,
        0, 1, 2, 3, 4, 5, 6, 7, 8, 9, 10, 11, 12, 13, 14, 15);
    const v8bf b1l = *(const v8bf*)&BsT[nbase + 16 + nc][kbB];
    const v8bf b1h = *(const v8bf*)&BsT[nbase + 16 + nc][kbB + 16];
    const v16bf bf1 = __builtin_shufflevector(b1l, b1h,
        0, 1, 2, 3, 4, 5, 6, 7, 8, 9, 10, 11, 12, 13, 14, 15);

    acc00 = __builtin_amdgcn_wmma_f32_16x16x32_bf16(false, af0, false, bf0, (short)0, acc00, false, false);
    acc01 = __builtin_amdgcn_wmma_f32_16x16x32_bf16(false, af0, false, bf1, (short)0, acc01, false, false);
    acc10 = __builtin_amdgcn_wmma_f32_16x16x32_bf16(false, af1, false, bf0, (short)0, acc10, false, false);
    acc11 = __builtin_amdgcn_wmma_f32_16x16x32_bf16(false, af1, false, bf1, (short)0, acc11, false, false);
    __syncthreads();
  }

  // epilogue: C layout — VGPR r: row = r (+8 for upper lane half), col = lane&15
  const int mbase = m0 + ((w & 3) << 5) + ((lane < 16) ? 0 : 8);
  const int nb    = n0 + ((w >> 2) << 5) + (lane & 15);
  #pragma unroll
  for (int mi = 0; mi < 2; ++mi) {
    #pragma unroll
    for (int half = 0; half < 2; ++half) {
      const int gn = nb + (half << 4);
      if (gn >= N) continue;
      const v8f& acc = mi ? (half ? acc11 : acc10) : (half ? acc01 : acc00);
      const float bsv = bb ? bb[gn] : 0.0f;
      #pragma unroll
      for (int r = 0; r < 8; ++r) {
        const int gm = mbase + (mi << 4) + r;
        if (gm >= M) continue;
        float val = acc[r] * scale + bsv;
        if (Db) {
          const long ar = (addRowDiv > 1) ? (long)(gm / addRowDiv) : (long)gm;
          const float av = Db[ar * (long)ldadd + gn];
          if (act == 3) { Cb[(long)gm * ldc + gn] = sigf(val) * av; continue; }
          val += av;
        }
        if (act == 1)      val = (val > 0.0f) ? val : (__expf(val) - 1.0f);
        else if (act == 2) val = sigf(val);
        else if (act == 3) val = sigf(val);
        Cb[(long)gm * ldc + gn] = val;
      }
    }
  }
}

// ---------------- elementwise / reduction kernels ----------------
__global__ void embed_k(const float* __restrict__ d, const float* __restrict__ W,
                        const float* __restrict__ b, float* __restrict__ out,
                        long BT, int N, int Hd)
{
  long idx = (long)blockIdx.x * blockDim.x + threadIdx.x;
  long tot = BT * N * Hd;
  if (idx >= tot) return;
  int h = (int)(idx % Hd);
  long tmp = idx / Hd;
  int n = (int)(tmp % N);
  long bt = tmp / N;
  out[idx] = d[bt * N + n] * W[(long)n * Hd + h] + b[(long)n * Hd + h];
}

__global__ void add_k(const float* __restrict__ a, const float* __restrict__ b,
                      float* __restrict__ o, long n)
{
  long i = (long)blockIdx.x * blockDim.x + threadIdx.x;
  if (i < n) o[i] = a[i] + b[i];
}

__global__ void zero_k(float* __restrict__ p, long n)
{
  long i = (long)blockIdx.x * blockDim.x + threadIdx.x;
  if (i < n) p[i] = 0.0f;
}

// wave-per-row LayerNorm (biased var, eps=1e-3); gamma row = (r % gmod)
__global__ void layernorm_k(const float* __restrict__ x, const float* __restrict__ g,
                            const float* __restrict__ bt, float* __restrict__ o,
                            long rows, int len, int gmod)
{
  long r = ((long)blockIdx.x * blockDim.x + threadIdx.x) >> 5;
  int lane = threadIdx.x & 31;
  if (r >= rows) return;
  const float* xr = x + r * len;
  float s = 0.0f, s2 = 0.0f;
  for (int i = lane; i < len; i += 32) { float v = xr[i]; s += v; s2 += v * v; }
  for (int off = 16; off; off >>= 1) { s += __shfl_xor(s, off, 32); s2 += __shfl_xor(s2, off, 32); }
  float mean = s / len;
  float var  = s2 / len - mean * mean;
  float rstd = rsqrtf(var + 1e-3f);
  long go = (long)(r % gmod) * len;
  float* orow = o + r * len;
  for (int i = lane; i < len; i += 32)
    orow[i] = (xr[i] - mean) * rstd * g[go + i] + bt[go + i];
}

__global__ void softmax_small_k(float* __restrict__ x, long rows, int len)
{
  long r = (long)blockIdx.x * blockDim.x + threadIdx.x;
  if (r >= rows) return;
  float* p = x + r * len;
  float mx = -1e30f;
  for (int i = 0; i < len; ++i) mx = fmaxf(mx, p[i]);
  float s = 0.0f;
  for (int i = 0; i < len; ++i) { float e = __expf(p[i] - mx); p[i] = e; s += e; }
  float inv = 1.0f / s;
  for (int i = 0; i < len; ++i) p[i] *= inv;
}

// out[b, tOff+t, h] = sum_n g[bt,n,h] * w[bt,n]
__global__ void wsum_k(const float* __restrict__ g, const float* __restrict__ w,
                       float* __restrict__ out, int Bc, int Tl, int N, int Hd,
                       int tOff, int Tfull)
{
  long idx = (long)blockIdx.x * blockDim.x + threadIdx.x;
  long tot = (long)Bc * Tl * Hd;
  if (idx >= tot) return;
  int h = (int)(idx % Hd);
  long bt = idx / Hd;
  int t = (int)(bt % Tl);
  int b = (int)(bt / Tl);
  float acc = 0.0f;
  for (int n = 0; n < N; ++n)
    acc += g[((long)bt * N + n) * Hd + h] * w[(long)bt * N + n];
  out[((long)b * Tfull + tOff + t) * (long)Hd + h] = acc;
}

// keras LSTM gates, order i,f,g,o; writes h into sequence buffer
__global__ void lstm_gates_k(const float* __restrict__ z, float* __restrict__ c,
                             float* __restrict__ h, float* __restrict__ seq,
                             int Bc, int Hn, long seqStride)
{
  int idx = blockIdx.x * blockDim.x + threadIdx.x;
  if (idx >= Bc * Hn) return;
  int b = idx / Hn, u = idx % Hn;
  const float* zr = z + (long)b * 4 * Hn;
  float ig = sigf(zr[u]);
  float fg = sigf(zr[Hn + u]);
  float gg = tanhf(zr[2 * Hn + u]);
  float og = sigf(zr[3 * Hn + u]);
  float cn = fg * c[idx] + ig * gg;
  float hn = og * tanhf(cn);
  c[idx] = cn; h[idx] = hn;
  seq[(long)b * seqStride + u] = hn;
}

// wave-per-row causal softmax over rows of length Tn; row r -> t = r % Tn
__global__ void attn_softmax_k(float* __restrict__ s, long rows, int Tn)
{
  long r = ((long)blockIdx.x * blockDim.x + threadIdx.x) >> 5;
  int lane = threadIdx.x & 31;
  if (r >= rows) return;
  int t = (int)(r % Tn);
  float* p = s + r * Tn;
  float mx = -1e30f;
  for (int i = lane; i <= t; i += 32) mx = fmaxf(mx, p[i]);
  for (int off = 16; off; off >>= 1) mx = fmaxf(mx, __shfl_xor(mx, off, 32));
  float sum = 0.0f;
  for (int i = lane; i <= t; i += 32) sum += __expf(p[i] - mx);
  for (int off = 16; off; off >>= 1) sum += __shfl_xor(sum, off, 32);
  float inv = 1.0f / sum;
  for (int i = lane; i < Tn; i += 32)
    p[i] = (i <= t) ? (__expf(p[i] - mx) * inv) : 0.0f;
}

// mean over heads: heads laid out [(head*B + b), t, k]
__global__ void meanheads_k(const float* __restrict__ hd, float* __restrict__ out,
                            int Bc, int Tn, int Dk, int Hh)
{
  long idx = (long)blockIdx.x * blockDim.x + threadIdx.x;
  long tot = (long)Bc * Tn * Dk;
  if (idx >= tot) return;
  int k = (int)(idx % Dk);
  long bt = idx / Dk;
  int b = (int)(bt / Tn);
  int t = (int)(bt % Tn);
  float a = 0.0f;
  for (int n = 0; n < Hh; ++n)
    a += hd[(((long)n * Bc + b) * Tn + t) * Dk + k];
  out[idx] = a / (float)Hh;
}

// ---------------- host side ----------------
struct GrnP { const float *W1,*b1,*W3,*b3,*Wa,*ba,*Wg,*bg,*lg,*lb,*Wc,*Ws,*bs; };

#define GRID1(n) dim3((unsigned)(((long)(n) + 255) / 256)), dim3(256), 0, stream

extern "C" void kernel_launch(void* const* d_in, const int* in_sizes, int n_in,
                              void* d_out, int out_size, void* d_ws, size_t ws_size,
                              hipStream_t stream)
{
  (void)in_sizes; (void)n_in; (void)out_size; (void)ws_size;

  // ---- read inputs/params in setup_inputs() insertion order (104 leaves) ----
  int pi = 0;
  auto rd = [&]() -> const float* { return (const float*)d_in[pi++]; };
  auto rdGrn = [&](bool ctx, bool skip) {
    GrnP p{};
    p.W1 = rd(); p.b1 = rd(); p.W3 = rd(); p.b3 = rd();
    p.Wa = rd(); p.ba = rd(); p.Wg = rd(); p.bg = rd();
    p.lg = rd(); p.lb = rd();
    if (ctx)  p.Wc = rd();
    if (skip) { p.Ws = rd(); p.bs = rd(); }
    return p;
  };

  const float* h_data = rd();
  const float* f_data = rd();
  const float* statv  = rd();
  const float* h_emb_W = rd(); const float* h_emb_b = rd();
  const float* f_emb_W = rd(); const float* f_emb_b = rd();
  GrnP hf   = rdGrn(false, false);      // h_feat (stacked, per-feature)
  GrnP ff   = rdGrn(false, false);      // f_feat
  GrnP hsel = rdGrn(true, true);        // h_sel
  GrnP fsel = rdGrn(true, true);        // f_sel
  const float* lh_Wk = rd(); const float* lh_Wr = rd(); const float* lh_b = rd();
  const float* lf_Wk = rd(); const float* lf_Wr = rd(); const float* lf_b = rd();
  const float* g1Wa = rd(); const float* g1ba = rd(); const float* g1Wg = rd(); const float* g1bg = rd();
  const float* ln1g = rd(); const float* ln1b = rd();
  GrnP enr = rdGrn(true, false);        // enrich
  const float* attWq = rd(); const float* attWk = rd();
  const float* attWv = rd(); const float* attWo = rd();
  const float* g2Wa = rd(); const float* g2ba = rd(); const float* g2Wg = rd(); const float* g2bg = rd();
  const float* ln2g = rd(); const float* ln2b = rd();
  GrnP decp = rdGrn(false, false);      // dec
  const float* g3Wa = rd(); const float* g3ba = rd(); const float* g3Wg = rd(); const float* g3bg = rd();
  const float* ln3g = rd(); const float* ln3b = rd();
  const float* W_out = rd(); const float* b_out = rd();
  float* out = (float*)d_out;

  // ---- workspace layout (bump allocator in floats) ----
  float* ws = (float*)d_ws;
  size_t off = 0;
  auto alloc = [&](long n) { float* p = ws + off; off += (size_t)n; return p; };

  const long nEh = kBTH * kNH * (long)kH;   // 88M floats
  const long nEf = kBTF * kNF * (long)kH;
  float* embH  = alloc(nEh);
  float* ghB   = alloc(nEh);
  float* embF  = alloc(nEf);
  float* gfB   = alloc(nEf);
  float* t1    = alloc(nEh);               // generic big temps
  float* t2    = alloc(nEh);
  float* t3    = alloc(nEh);
  float* inp      = alloc(kBT * (long)kH);
  float* lstmO    = alloc(kBT * (long)kH);
  float* temporal = alloc(kBT * (long)kH);
  float* enriched = alloc(kBT * (long)kH);
  float* xbuf     = alloc(kBT * (long)kH);
  float* decoded  = alloc(kBT * (long)kH);
  float* tfd      = alloc(kBT * (long)kH);
  float* xsH   = alloc((long)kB * kTH * 4 * kH);
  float* xsF   = alloc((long)kB * kTF * 4 * kH);
  float* cst   = alloc((long)kB * kH);
  float* hst   = alloc((long)kB * kH);
  float* zbuf  = alloc((long)kB * 4 * kH);
  float* cproj = alloc((long)kB * kH);
  float* whB   = alloc(kBTH * kNH);
  float* wfB   = alloc(kBTF * kNF);
  float* selT  = alloc(kBTH * 8);
  float* selT2 = alloc(kBTH * 8);
  float* qB    = alloc((long)kHEADS * kBT * kDK);
  float* kBuf  = alloc((long)kHEADS * kBT * kDK);
  float* vB    = alloc(kBT * (long)kDK);
  float* sc    = alloc((long)kHEADS * kB * kT * kT);
  float* hdB   = alloc((long)kHEADS * kBT * kDK);
  float* mh    = alloc(kBT * (long)kDK);

  // ---- GEMM launcher with defaults (M-tile 128, N-tile 64) ----
  auto GEMM = [&](const float* A, const float* Bm, const float* bias, float* C,
                  int M, int N, int K, int lda, int ldb, int ldc,
                  int batches = 1, long sA = 0, long sB = 0, long sC = 0,
                  int act = 0, float scale = 1.0f, int transB = 0,
                  const float* Add = nullptr, int ldadd = 0, long sAdd = 0,
                  int addRowDiv = 1, int addMod = 0, int aMod = 0, int bMod = 0,
                  int biasStride = 0) {
    dim3 grid((unsigned)((N + 63) / 64), (unsigned)((M + 127) / 128), (unsigned)batches);
    gemm_wmma<<<grid, dim3(256), 0, stream>>>(A, Bm, bias, Add, C,
        M, N, K, lda, ldb, ldc, ldadd, sA, sB, sC, sAdd,
        aMod, bMod, addMod, biasStride, transB, act, scale, addRowDiv);
  };

  const int H = kH, NHH = kNH * kH, NFH = kNF * kH;

  // ---- 1) embeddings ----
  embed_k<<<GRID1(nEh)>>>(h_data, h_emb_W, h_emb_b, embH, kBTH, kNH, H);
  embed_k<<<GRID1(nEf)>>>(f_data, f_emb_W, f_emb_b, embF, kBTF, kNF, H);

  // ---- 2) per-feature GRN stacks (batched over feature n) ----
  // h stack
  GEMM(embH, hf.W1, hf.b1, t1, (int)kBTH, H, H, NHH, H, NHH,
       kNH, (long)H, (long)H * H, (long)H, 1, 1.0f, 0, nullptr, 0, 0, 1, 0, 0, 0, H);
  GEMM(t1, hf.W3, hf.b3, t2, (int)kBTH, H, H, NHH, H, NHH,
       kNH, (long)H, (long)H * H, (long)H, 0, 1.0f, 0, nullptr, 0, 0, 1, 0, 0, 0, H);
  GEMM(t2, hf.Wa, hf.ba, t1, (int)kBTH, H, H, NHH, H, NHH,
       kNH, (long)H, (long)H * H, (long)H, 0, 1.0f, 0, nullptr, 0, 0, 1, 0, 0, 0, H);
  GEMM(t2, hf.Wg, hf.bg, t3, (int)kBTH, H, H, NHH, H, NHH,
       kNH, (long)H, (long)H * H, (long)H, 3, 1.0f, 0, t1, NHH, (long)H, 1, 0, 0, 0, H);
  add_k<<<GRID1(nEh)>>>(embH, t3, ghB, nEh);
  layernorm_k<<<GRID1(kBTH * kNH * 32)>>>(ghB, hf.lg, hf.lb, ghB, kBTH * kNH, H, kNH);
  // f stack
  GEMM(embF, ff.W1, ff.b1, t1, (int)kBTF, H, H, NFH, H, NFH,
       kNF, (long)H, (long)H * H, (long)H, 1, 1.0f, 0, nullptr, 0, 0, 1, 0, 0, 0, H);
  GEMM(t1, ff.W3, ff.b3, t2, (int)kBTF, H, H, NFH, H, NFH,
       kNF, (long)H, (long)H * H, (long)H, 0, 1.0f, 0, nullptr, 0, 0, 1, 0, 0, 0, H);
  GEMM(t2, ff.Wa, ff.ba, t1, (int)kBTF, H, H, NFH, H, NFH,
       kNF, (long)H, (long)H * H, (long)H, 0, 1.0f, 0, nullptr, 0, 0, 1, 0, 0, 0, H);
  GEMM(t2, ff.Wg, ff.bg, t3, (int)kBTF, H, H, NFH, H, NFH,
       kNF, (long)H, (long)H * H, (long)H, 3, 1.0f, 0, t1, NFH, (long)H, 1, 0, 0, 0, H);
  add_k<<<GRID1(nEf)>>>(embF, t3, gfB, nEf);
  layernorm_k<<<GRID1(kBTF * kNF * 32)>>>(gfB, ff.lg, ff.lb, gfB, kBTF * kNF, H, kNF);

  // ---- 3) variable selection GRNs + softmax + weighted sum ----
  // h
  GEMM(statv, hsel.Wc, nullptr, cproj, kB, H, H, H, H, H);
  GEMM(embH, hsel.W1, hsel.b1, t1, (int)kBTH, H, NHH, NHH, H, H,
       1, 0, 0, 0, 1, 1.0f, 0, cproj, H, 0, kTH);
  GEMM(t1, hsel.W3, hsel.b3, t2, (int)kBTH, H, H, H, H, H);
  GEMM(t2, hsel.Wa, hsel.ba, selT, (int)kBTH, kNH, H, H, kNH, kNH);
  GEMM(t2, hsel.Wg, hsel.bg, selT2, (int)kBTH, kNH, H, H, kNH, kNH,
       1, 0, 0, 0, 3, 1.0f, 0, selT, kNH, 0, 1);
  GEMM(embH, hsel.Ws, hsel.bs, selT, (int)kBTH, kNH, NHH, NHH, kNH, kNH);
  add_k<<<GRID1(kBTH * kNH)>>>(selT, selT2, whB, kBTH * kNH);
  layernorm_k<<<GRID1(kBTH * 32)>>>(whB, hsel.lg, hsel.lb, whB, kBTH, kNH, 1);
  softmax_small_k<<<GRID1(kBTH)>>>(whB, kBTH, kNH);
  wsum_k<<<GRID1(kBTH * (long)H)>>>(ghB, whB, inp, kB, kTH, kNH, H, 0, kT);
  // f
  GEMM(statv, fsel.Wc, nullptr, cproj, kB, H, H, H, H, H);
  GEMM(embF, fsel.W1, fsel.b1, t1, (int)kBTF, H, NFH, NFH, H, H,
       1, 0, 0, 0, 1, 1.0f, 0, cproj, H, 0, kTF);
  GEMM(t1, fsel.W3, fsel.b3, t2, (int)kBTF, H, H, H, H, H);
  GEMM(t2, fsel.Wa, fsel.ba, selT, (int)kBTF, kNF, H, H, kNF, kNF);
  GEMM(t2, fsel.Wg, fsel.bg, selT2, (int)kBTF, kNF, H, H, kNF, kNF,
       1, 0, 0, 0, 3, 1.0f, 0, selT, kNF, 0, 1);
  GEMM(embF, fsel.Ws, fsel.bs, selT, (int)kBTF, kNF, NFH, NFH, kNF, kNF);
  add_k<<<GRID1(kBTF * kNF)>>>(selT, selT2, wfB, kBTF * kNF);
  layernorm_k<<<GRID1(kBTF * 32)>>>(wfB, fsel.lg, fsel.lb, wfB, kBTF, kNF, 1);
  softmax_small_k<<<GRID1(kBTF)>>>(wfB, kBTF, kNF);
  wsum_k<<<GRID1(kBTF * (long)H)>>>(gfB, wfB, inp, kB, kTF, kNF, H, kTH, kT);

  // ---- 4) LSTM encoder/decoder ----
  GEMM(inp, lh_Wk, lh_b, xsH, kTH, 4 * H, H, H, 4 * H, 4 * H,
       kB, (long)kT * H, 0, (long)kTH * 4 * H);
  GEMM(inp + (long)kTH * H, lf_Wk, lf_b, xsF, kTF, 4 * H, H, H, 4 * H, 4 * H,
       kB, (long)kT * H, 0, (long)kTF * 4 * H);
  zero_k<<<GRID1(kB * H)>>>(cst, (long)kB * H);
  zero_k<<<GRID1(kB * H)>>>(hst, (long)kB * H);
  for (int t = 0; t < kTH; ++t) {
    GEMM(hst, lh_Wr, nullptr, zbuf, kB, 4 * H, H, H, 4 * H, 4 * H,
         1, 0, 0, 0, 0, 1.0f, 0, xsH + (long)t * 4 * H, kTH * 4 * H, 0, 1);
    lstm_gates_k<<<GRID1(kB * H)>>>(zbuf, cst, hst, lstmO + (long)t * H, kB, H, (long)kT * H);
  }
  for (int t = 0; t < kTF; ++t) {
    GEMM(hst, lf_Wr, nullptr, zbuf, kB, 4 * H, H, H, 4 * H, 4 * H,
         1, 0, 0, 0, 0, 1.0f, 0, xsF + (long)t * 4 * H, kTF * 4 * H, 0, 1);
    lstm_gates_k<<<GRID1(kB * H)>>>(zbuf, cst, hst, lstmO + (long)(kTH + t) * H, kB, H, (long)kT * H);
  }

  // ---- 5) temporal = LN(inp + GLU(lstm_out)) ----
  GEMM(lstmO, g1Wa, g1ba, t1, (int)kBT, H, H, H, H, H);
  GEMM(lstmO, g1Wg, g1bg, t2, (int)kBT, H, H, H, H, H,
       1, 0, 0, 0, 3, 1.0f, 0, t1, H, 0, 1);
  add_k<<<GRID1(kBT * (long)H)>>>(inp, t2, t3, kBT * (long)H);
  layernorm_k<<<GRID1(kBT * 32)>>>(t3, ln1g, ln1b, temporal, kBT, H, 1);

  // ---- 6) static enrichment GRN ----
  GEMM(statv, enr.Wc, nullptr, cproj, kB, H, H, H, H, H);
  GEMM(temporal, enr.W1, enr.b1, t1, (int)kBT, H, H, H, H, H,
       1, 0, 0, 0, 1, 1.0f, 0, cproj, H, 0, kT);
  GEMM(t1, enr.W3, enr.b3, t2, (int)kBT, H, H, H, H, H);
  GEMM(t2, enr.Wa, enr.ba, t1, (int)kBT, H, H, H, H, H);
  GEMM(t2, enr.Wg, enr.bg, t3, (int)kBT, H, H, H, H, H,
       1, 0, 0, 0, 3, 1.0f, 0, t1, H, 0, 1);
  add_k<<<GRID1(kBT * (long)H)>>>(temporal, t3, t1, kBT * (long)H);
  layernorm_k<<<GRID1(kBT * 32)>>>(t1, enr.lg, enr.lb, enriched, kBT, H, 1);

  // ---- 7) interpretable MHA with causal mask ----
  GEMM(enriched, attWq, nullptr, qB, (int)kBT, kDK, H, H, kDK, kDK,
       kHEADS, 0, (long)H * kDK, kBT * (long)kDK);
  GEMM(enriched, attWk, nullptr, kBuf, (int)kBT, kDK, H, H, kDK, kDK,
       kHEADS, 0, (long)H * kDK, kBT * (long)kDK);
  GEMM(enriched, attWv, nullptr, vB, (int)kBT, kDK, H, H, kDK, kDK);
  // scores[z] = q[z] @ k[z]^T / sqrt(DK);  z = head*B + b
  GEMM(qB, kBuf, nullptr, sc, kT, kT, kDK, kDK, kDK, kT,
       kHEADS * kB, (long)kT * kDK, (long)kT * kDK, (long)kT * kT, 0, 0.125f, 1);
  attn_softmax_k<<<GRID1((long)kHEADS * kB * kT * 32)>>>(sc, (long)kHEADS * kB * kT, kT);
  // heads[z] = attn[z] @ v[b]  (bMod = B picks b = z % B)
  GEMM(sc, vB, nullptr, hdB, kT, kDK, kT, kT, kDK, kDK,
       kHEADS * kB, (long)kT * kT, (long)kT * kDK, (long)kT * kDK, 0, 1.0f, 0,
       nullptr, 0, 0, 1, 0, 0, kB);
  meanheads_k<<<GRID1(kBT * (long)kDK)>>>(hdB, mh, kB, kT, kDK, kHEADS);
  GEMM(mh, attWo, nullptr, t1, (int)kBT, H, kDK, kDK, H, H);

  // ---- 8) x = LN(enriched + GLU(attn_out)) ----
  GEMM(t1, g2Wa, g2ba, t2, (int)kBT, H, H, H, H, H);
  GEMM(t1, g2Wg, g2bg, t3, (int)kBT, H, H, H, H, H,
       1, 0, 0, 0, 3, 1.0f, 0, t2, H, 0, 1);
  add_k<<<GRID1(kBT * (long)H)>>>(enriched, t3, t2, kBT * (long)H);
  layernorm_k<<<GRID1(kBT * 32)>>>(t2, ln2g, ln2b, xbuf, kBT, H, 1);

  // ---- 9) position-wise decoder GRN ----
  GEMM(xbuf, decp.W1, decp.b1, t1, (int)kBT, H, H, H, H, H, 1, 0, 0, 0, 1);
  GEMM(t1, decp.W3, decp.b3, t2, (int)kBT, H, H, H, H, H);
  GEMM(t2, decp.Wa, decp.ba, t1, (int)kBT, H, H, H, H, H);
  GEMM(t2, decp.Wg, decp.bg, t3, (int)kBT, H, H, H, H, H,
       1, 0, 0, 0, 3, 1.0f, 0, t1, H, 0, 1);
  add_k<<<GRID1(kBT * (long)H)>>>(xbuf, t3, t1, kBT * (long)H);
  layernorm_k<<<GRID1(kBT * 32)>>>(t1, decp.lg, decp.lb, decoded, kBT, H, 1);

  // ---- 10) tfd = LN(temporal + GLU(decoded)) ----
  GEMM(decoded, g3Wa, g3ba, t1, (int)kBT, H, H, H, H, H);
  GEMM(decoded, g3Wg, g3bg, t2, (int)kBT, H, H, H, H, H,
       1, 0, 0, 0, 3, 1.0f, 0, t1, H, 0, 1);
  add_k<<<GRID1(kBT * (long)H)>>>(temporal, t2, t1, kBT * (long)H);
  layernorm_k<<<GRID1(kBT * 32)>>>(t1, ln3g, ln3b, tfd, kBT, H, 1);

  // ---- 11) quantile head on forecast horizon: tfd[:, TH:, :] @ W_out + b_out ----
  GEMM(tfd + (long)kTH * H, W_out, b_out, out, kTF, 3, H, H, 3, 3,
       kB, (long)kT * H, 0, (long)kTF * 3);
}
